// DepDagLSTMPool_6038724018710
// MI455X (gfx1250) — matrix operand order repaired and checked by reading, not compiled, not run on hardware
//
#include <hip/hip_runtime.h>
#include <hip/hip_bf16.h>

// ---------------- problem constants (from reference) ----------------
#define NN     100000
#define DD     256
#define ATT    512
#define HH     8
#define HD     64
#define EE     250000
#define TS     64
#define PP     8192
#define PCC    1024
#define MPP    131072
#define MCC    16384
#define CH     768          // D + ATT

typedef __attribute__((ext_vector_type(16))) __bf16 v16bf;
typedef __attribute__((ext_vector_type(8)))  float  v8f;
typedef __attribute__((ext_vector_type(4)))  unsigned int v4u;
typedef __attribute__((ext_vector_type(8)))  int    v8i;
typedef __attribute__((ext_vector_type(4)))  int    v4i;

#if defined(__has_builtin)
#if __has_builtin(__builtin_amdgcn_tensor_load_to_lds) && __has_builtin(__builtin_amdgcn_s_wait_tensorcnt)
#define HAVE_TDM 1
#endif
#endif

// ---------------- helpers ----------------
__device__ __forceinline__ void atomicMaxF(float* addr, float val) {
    if (val >= 0.0f) atomicMax((int*)addr, __float_as_int(val));
    else             atomicMin((unsigned int*)addr, (unsigned int)__float_as_int(val));
}

__global__ void fill_f32(float* __restrict__ p, float v, long n) {
    long i = (long)blockIdx.x * blockDim.x + threadIdx.x;
    if (i < n) p[i] = v;
}

// ---------------- weight prep: f32 -> bf16 (transposed) ----------------
__global__ void prep_qkvT(const float* __restrict__ W, __bf16* __restrict__ out) {
    int idx = blockIdx.x * blockDim.x + threadIdx.x;     // over ATT*DD
    if (idx >= ATT * DD) return;
    int j = idx / DD, d = idx % DD;
    out[idx] = (__bf16)W[(j >> 6) * (DD * HD) + d * HD + (j & 63)];
}
__global__ void prep_T(const float* __restrict__ W, __bf16* __restrict__ out, int Kdim, int Ncols) {
    long idx = (long)blockIdx.x * blockDim.x + threadIdx.x;
    if (idx >= (long)Kdim * Ncols) return;
    int j = (int)(idx / Kdim), d = (int)(idx % Kdim);
    out[idx] = (__bf16)W[(long)d * Ncols + j];
}
__global__ void prep_cvt(const float* __restrict__ W, __bf16* __restrict__ out, long n) {
    long i = (long)blockIdx.x * blockDim.x + threadIdx.x;
    if (i < n) out[i] = (__bf16)W[i];
}

// ---------------- layernorm: one wave per row (wave32) ----------------
__global__ void layernorm_kernel(const float* __restrict__ xin,
                                 const float* __restrict__ g, const float* __restrict__ b,
                                 float* __restrict__ xout, __bf16* __restrict__ xb) {
    int row = blockIdx.x * (blockDim.x >> 5) + (threadIdx.x >> 5);
    if (row >= NN) return;
    int lane = threadIdx.x & 31;
    const float* p = xin + (long)row * DD;
    float vals[8]; float s = 0.f;
#pragma unroll
    for (int i = 0; i < 8; ++i) { vals[i] = p[lane + 32 * i]; s += vals[i]; }
#pragma unroll
    for (int off = 16; off; off >>= 1) s += __shfl_xor(s, off, 32);
    float mu = s * (1.0f / DD);
    float vs = 0.f;
#pragma unroll
    for (int i = 0; i < 8; ++i) { float d = vals[i] - mu; vs += d * d; }
#pragma unroll
    for (int off = 16; off; off >>= 1) vs += __shfl_xor(vs, off, 32);
    float rstd = rsqrtf(vs * (1.0f / DD) + 1e-5f);
#pragma unroll
    for (int i = 0; i < 8; ++i) {
        int c = lane + 32 * i;
        float y = (vals[i] - mu) * rstd * g[c] + b[c];
        xout[(long)row * DD + c] = y;
        xb[(long)row * DD + c]   = (__bf16)y;
    }
}

// ---------------- WMMA bf16 GEMM ----------------
// One block (8 waves) per 32-row tile; waves sweep 64-col groups of BT.
// A tile (32 x K, contiguous in memory) staged into LDS via the Tensor Data
// Mover; each B fragment feeds two A fragments (8 WMMAs) -> 32 FLOP per L2 byte.
// K/NC are compile-time so the 4 per-nt B rows sit at constant byte offsets
// from ONE base address (fits the 24-bit VMEM immediate) -> no address spills.
// EPI: 1 sigmoid(acc+bias), 2 out=gate*acc, 3 out+=gate*acc, 4 tanh(acc+bias), 5 bf16 store.

template <int K>
__device__ __forceinline__ void load_bfrag(v16bf* bn, const __bf16* __restrict__ bbase, int kk) {
    // bbase = BT + (g*64 + r)*K + half*16 ; row nt*16 away => constant offset nt*16*K
#pragma unroll
    for (int nt = 0; nt < 4; ++nt) {
        const __bf16* bp = bbase + nt * 16 * K + kk;
#pragma unroll
        for (int i = 0; i < 16; ++i) bn[nt][i] = bp[i];
    }
}

__device__ __forceinline__ v16bf load_afrag(const __bf16* base, int kk, int half) {
    // contiguous K order (matches B): lane<16 -> K 0..15, lane>=16 -> K 16..31
    v16bf a;
    const __bf16* ap = base + kk + half * 16;
#pragma unroll
    for (int i = 0; i < 8; ++i) { a[i] = ap[i]; a[8 + i] = ap[8 + i]; }
    return a;
}

template <int EPI, bool GATHER, int K, int NC>
__global__ __launch_bounds__(256, 1)
void wmma_gemm_bf16(const __bf16* __restrict__ A, const __bf16* __restrict__ BT,
                    float* __restrict__ outF, __bf16* __restrict__ outH,
                    const float* __restrict__ bias, const float* __restrict__ gate,
                    const int* __restrict__ rowidx, int M) {
    extern __shared__ __bf16 As[];                 // 32 * K bf16 (unused when GATHER)
    const int tile_m = blockIdx.x;
    if (tile_m * 32 >= M) return;
    const int lane = threadIdx.x & 31;
    const int wid  = threadIdx.x >> 5;
    const int r = lane & 15, half = lane >> 4;
    constexpr int ngrp = NC / 64;

    if (!GATHER) {
        // ---- stage contiguous 32xK A tile into LDS ----
#ifdef HAVE_TDM
        if (wid == 0) {
            unsigned long long ga = (unsigned long long)(const void*)(A + (long)tile_m * 32 * K);
            unsigned int lds = (unsigned int)(unsigned long long)(const void*)&As[0];
            // D# group 0: count=1 (valid), lds_addr, global_addr, type=2 ("image")
            v4u g0;
            g0[0] = 1u;
            g0[1] = lds;
            g0[2] = (unsigned int)ga;
            g0[3] = (unsigned int)(ga >> 32) | (2u << 30);
            // D# group 1: data_size=2B, tensor 32 x K (dim0=K), tile 32 x K, stride K
            v8i g1;
            g1[0] = (int)(1u << 16);                        // data_size = 1 -> 2 bytes
            g1[1] = (int)(((unsigned)K & 0xFFFFu) << 16);   // tensor_dim0[15:0] (bits 63:48)
            g1[2] = (int)((32u & 0xFFFFu) << 16);           // tensor_dim1[15:0]=32 (bits 95:80)
            g1[3] = (int)(((unsigned)K & 0xFFFFu) << 16);   // tile_dim0=K (bits 127:112)
            g1[4] = (int)(32u);                             // tile_dim1=32 (bits 143:128)
            g1[5] = (int)((unsigned)K);                     // tensor_dim0_stride[31:0]
            g1[6] = 0;
            g1[7] = 0;
            v4i z4; z4[0] = 0; z4[1] = 0; z4[2] = 0; z4[3] = 0;
#if __clang_major__ >= 23
            v8i z8; for (int i = 0; i < 8; ++i) z8[i] = 0;
            __builtin_amdgcn_tensor_load_to_lds(g0, g1, z4, z4, z8, 0);
#else
            __builtin_amdgcn_tensor_load_to_lds(g0, g1, z4, z4, 0);
#endif
            __builtin_amdgcn_s_wait_tensorcnt(0);
        }
        __syncthreads();
#else
        // fallback: cooperative copy (tile is one contiguous run of 32*K bf16)
        {
            const int chunks = (32 * K) / 8;               // 16B chunks
            const ulonglong2* srcp = (const ulonglong2*)(A + (long)tile_m * 32 * K);
            ulonglong2* dstp = (ulonglong2*)&As[0];
            for (int i = threadIdx.x; i < chunks; i += blockDim.x) dstp[i] = srcp[i];
        }
        __syncthreads();
#endif
    }

    // row base pointers for the two 16-row sub-tiles
    const __bf16* arow0 = GATHER ? (A + (long)rowidx[tile_m * 32 + r] * K)
                                 : (const __bf16*)&As[(long)r * K];
    const __bf16* arow1 = GATHER ? (A + (long)rowidx[tile_m * 32 + 16 + r] * K)
                                 : (const __bf16*)&As[(long)(16 + r) * K];

    for (int g = wid; g < ngrp; g += 8) {
        const __bf16* bbase = BT + (long)(g * 64 + r) * K + half * 16;

        v8f acc[2][4];
#pragma unroll
        for (int mt = 0; mt < 2; ++mt)
#pragma unroll
            for (int t = 0; t < 4; ++t)
#pragma unroll
                for (int i = 0; i < 8; ++i) acc[mt][t][i] = 0.0f;

        v16bf bA[4], bB[4];
        load_bfrag<K>(bA, bbase, 0);

        for (int kk = 0; kk < K; kk += 64) {
            // ---- step 1: consume bA (K-slice kk), prefetch bB (kk+32) ----
            if (kk + 32 < K) load_bfrag<K>(bB, bbase, kk + 32);
            {
                v16bf a0 = load_afrag(arow0, kk, half);
                v16bf a1 = load_afrag(arow1, kk, half);
#pragma unroll
                for (int nt = 0; nt < 4; ++nt)
                    acc[0][nt] = __builtin_amdgcn_wmma_f32_16x16x32_bf16(
                        false, a0, false, bA[nt], (short)0, acc[0][nt], false, false);
#pragma unroll
                for (int nt = 0; nt < 4; ++nt)
                    acc[1][nt] = __builtin_amdgcn_wmma_f32_16x16x32_bf16(
                        false, a1, false, bA[nt], (short)0, acc[1][nt], false, false);
            }
            // ---- step 2: consume bB (kk+32), prefetch bA (kk+64) ----
            if (kk + 64 < K) load_bfrag<K>(bA, bbase, kk + 64);
            {
                v16bf a0 = load_afrag(arow0, kk + 32, half);
                v16bf a1 = load_afrag(arow1, kk + 32, half);
#pragma unroll
                for (int nt = 0; nt < 4; ++nt)
                    acc[0][nt] = __builtin_amdgcn_wmma_f32_16x16x32_bf16(
                        false, a0, false, bB[nt], (short)0, acc[0][nt], false, false);
#pragma unroll
                for (int nt = 0; nt < 4; ++nt)
                    acc[1][nt] = __builtin_amdgcn_wmma_f32_16x16x32_bf16(
                        false, a1, false, bB[nt], (short)0, acc[1][nt], false, false);
            }
        }

        // C/D layout: VGPR i -> row = half*8 + i, col = lane&15
#pragma unroll
        for (int mt = 0; mt < 2; ++mt) {
#pragma unroll
            for (int nt = 0; nt < 4; ++nt) {
#pragma unroll
                for (int i = 0; i < 8; ++i) {
                    int row = tile_m * 32 + mt * 16 + half * 8 + i;
                    int col = g * 64 + nt * 16 + r;
                    long o = (long)row * NC + col;
                    float v = acc[mt][nt][i];
                    if (EPI == 1)      outF[o] = 1.0f / (1.0f + __expf(-(v + bias[col])));
                    else if (EPI == 2) outF[o] = gate[o] * v;
                    else if (EPI == 3) outF[o] += gate[o] * v;
                    else if (EPI == 4) outF[o] = tanhf(v + bias[col]);
                    else               outH[o] = (__bf16)v;
                }
            }
        }
    }
}

// ---------------- edge kernels (irregular part) ----------------
__global__ void score_kernel(const __bf16* __restrict__ Q, const __bf16* __restrict__ Km,
                             const int* __restrict__ src, const int* __restrict__ tgt,
                             float* __restrict__ sc, float* __restrict__ m) {
    long idx = (long)blockIdx.x * blockDim.x + threadIdx.x;
    if (idx >= (long)EE * HH) return;
    int e = (int)(idx >> 3), h = (int)(idx & 7);
    int s = src[e], t = tgt[e];
    const __bf16* q = Q  + (long)s * ATT + h * HD;
    const __bf16* k = Km + (long)t * ATT + h * HD;
    float acc = 0.f;
#pragma unroll 8
    for (int i = 0; i < HD; ++i) acc += (float)q[i] * (float)k[i];
    acc *= 0.125f;                                // 1/sqrt(64)
    sc[idx] = acc;
    atomicMaxF(&m[(long)s * HH + h], acc);
}

__global__ void exp_kernel(const int* __restrict__ src, float* __restrict__ sc,
                           const float* __restrict__ m, float* __restrict__ denom) {
    long idx = (long)blockIdx.x * blockDim.x + threadIdx.x;
    if (idx >= (long)EE * HH) return;
    int e = (int)(idx >> 3), h = (int)(idx & 7);
    float ex = __expf(sc[idx] - m[(long)src[e] * HH + h]);
    sc[idx] = ex;
    atomicAdd(&denom[(long)src[e] * HH + h], ex);
}

__global__ void agg_kernel(const int* __restrict__ src, const int* __restrict__ tgt,
                           const float* __restrict__ ex, const float* __restrict__ denom,
                           const __bf16* __restrict__ V, float* __restrict__ agg) {
    long idx = (long)blockIdx.x * blockDim.x + threadIdx.x;
    if (idx >= (long)EE * ATT) return;
    int e = (int)(idx >> 9), c = (int)(idx & 511);
    int h = c >> 6;
    int s = src[e], t = tgt[e];
    float prob = ex[(long)e * HH + h] / denom[(long)s * HH + h];
    float v = (float)V[(long)t * ATT + c];
    atomicAdd(&agg[(long)s * ATT + c], prob * v);
}

__global__ void cvt_f2b(const float* __restrict__ in, __bf16* __restrict__ out, long n) {
    long i = (long)blockIdx.x * blockDim.x + threadIdx.x;
    if (i < n) out[i] = (__bf16)in[i];
}

// h = [x | comb] as bf16
__global__ void build_h(const float* __restrict__ x, const float* __restrict__ comb,
                        __bf16* __restrict__ hb) {
    long idx = (long)blockIdx.x * blockDim.x + threadIdx.x;
    if (idx >= (long)NN * CH) return;
    int row = (int)(idx / CH), j = (int)(idx % CH);
    float v = (j < DD) ? x[(long)row * DD + j] : comb[(long)row * ATT + (j - DD)];
    hb[idx] = (__bf16)v;
}

// segment-max pooling: out[seg*stride + j] = max(pooled[node, j])
__global__ void pool_max(const float* __restrict__ pooled, const int* __restrict__ nodes,
                         const int* __restrict__ segs, float* __restrict__ out,
                         int stride, long count) {
    long idx = (long)blockIdx.x * blockDim.x + threadIdx.x;
    if (idx >= count * CH) return;
    int i = (int)(idx / CH), j = (int)(idx % CH);
    float v = pooled[(long)nodes[i] * CH + j];
    atomicMaxF(&out[(long)segs[i] * stride + j], v);
}

__global__ void gather_conj(const float* __restrict__ conj_sets, const int* __restrict__ p2c,
                            float* __restrict__ out) {
    long idx = (long)blockIdx.x * blockDim.x + threadIdx.x;
    if (idx >= (long)PP * CH) return;
    int p = (int)(idx / CH), j = (int)(idx % CH);
    out[(long)p * (2 * CH) + CH + j] = conj_sets[(long)p2c[p] * CH + j];
}

// ---------------- host side ----------------
static inline int cdiv(long a, long b) { return (int)((a + b - 1) / b); }

extern "C" void kernel_launch(void* const* d_in, const int* in_sizes, int n_in,
                              void* d_out, int out_size, void* d_ws, size_t ws_size,
                              hipStream_t stream) {
    // ---- inputs ----
    const float* node_states = (const float*)d_in[0];
    const int*   node_types  = (const int*)d_in[1];
    const int*   pair_conj   = (const int*)d_in[2];
    const int*   pair_prem   = (const int*)d_in[3];
    const int*   pool_p_nod  = (const int*)d_in[4];
    const int*   pool_p_seg  = (const int*)d_in[5];
    const int*   pool_c_nod  = (const int*)d_in[6];
    const int*   pool_c_seg  = (const int*)d_in[7];
    const int*   prem2conj   = (const int*)d_in[8];
    const float* ln_g  = (const float*)d_in[9];
    const float* ln_b  = (const float*)d_in[10];
    const float* emb_p = (const float*)d_in[11];
    const float* wp_w  = (const float*)d_in[12];
    const float* wp_b  = (const float*)d_in[13];
    const float* emb_c = (const float*)d_in[14];
    const float* wc_w  = (const float*)d_in[15];
    const float* wc_b  = (const float*)d_in[16];
    const float* Wq_p  = (const float*)d_in[17];
    const float* Wk_p  = (const float*)d_in[18];
    const float* Wv_p  = (const float*)d_in[19];
    const float* Wo_p  = (const float*)d_in[20];
    const float* Wq_c  = (const float*)d_in[21];
    const float* Wk_c  = (const float*)d_in[22];
    const float* Wv_c  = (const float*)d_in[23];
    const float* Wo_c  = (const float*)d_in[24];
    const float* W_dag = (const float*)d_in[25];
    const float* b_dag = (const float*)d_in[26];
    float* out = (float*)d_out;

    // ---- workspace carve ----
    char* base = (char*)d_ws;
    size_t off = 0;
    auto carve = [&](size_t bytes) -> char* {
        off = (off + 255) & ~(size_t)255;
        char* p = base + off;
        off += bytes;
        return p;
    };
    // persistent
    float*  x    = (float*) carve((size_t)NN * DD * 4);
    __bf16* xb   = (__bf16*)carve((size_t)NN * DD * 2);
    float*  comb = (float*) carve((size_t)NN * ATT * 4);
    __bf16* wqT[2], *wkT[2], *wvT[2], *woT[2];
    for (int d = 0; d < 2; ++d) {
        wqT[d] = (__bf16*)carve((size_t)ATT * DD * 2);
        wkT[d] = (__bf16*)carve((size_t)ATT * DD * 2);
        wvT[d] = (__bf16*)carve((size_t)ATT * DD * 2);
        woT[d] = (__bf16*)carve((size_t)ATT * ATT * 2);
    }
    __bf16* wdagT = (__bf16*)carve((size_t)CH * CH * 2);
    __bf16* wgT[2]; __bf16* embb[2];
    for (int d = 0; d < 2; ++d) {
        wgT[d]  = (__bf16*)carve((size_t)ATT * TS * 2);
        embb[d] = (__bf16*)carve((size_t)TS * TS * 2);
    }
    // phase region (phase-2 overlays phase-1)
    size_t phase_off = (off + 255) & ~(size_t)255;
    // phase 1
    off = phase_off;
    __bf16* Qb   = (__bf16*)carve((size_t)NN * ATT * 2);
    __bf16* Kb   = (__bf16*)carve((size_t)NN * ATT * 2);
    __bf16* Vb   = (__bf16*)carve((size_t)NN * ATT * 2);
    float*  gatebuf = (float*)carve((size_t)NN * ATT * 4);
    float*  agg  = (float*) carve((size_t)NN * ATT * 4);
    __bf16* aggb = (__bf16*)carve((size_t)NN * ATT * 2);
    float*  sc   = (float*) carve((size_t)EE * HH * 4);
    float*  mbuf = (float*) carve((size_t)NN * HH * 4);
    float*  den  = (float*) carve((size_t)NN * HH * 4);
    // phase 2 (reuse)
    off = phase_off;
    __bf16* hb     = (__bf16*)carve((size_t)NN * CH * 2);
    float*  pooled = (float*) carve((size_t)NN * CH * 4);
    float*  csets  = (float*) carve((size_t)PCC * CH * 4);
    (void)ws_size; (void)n_in; (void)in_sizes; (void)out_size;

    const int TB = 256;
    // ---- weight prep ----
    prep_qkvT<<<cdiv((long)ATT * DD, TB), TB, 0, stream>>>(Wq_p, wqT[0]);
    prep_qkvT<<<cdiv((long)ATT * DD, TB), TB, 0, stream>>>(Wk_p, wkT[0]);
    prep_qkvT<<<cdiv((long)ATT * DD, TB), TB, 0, stream>>>(Wv_p, wvT[0]);
    prep_qkvT<<<cdiv((long)ATT * DD, TB), TB, 0, stream>>>(Wq_c, wqT[1]);
    prep_qkvT<<<cdiv((long)ATT * DD, TB), TB, 0, stream>>>(Wk_c, wkT[1]);
    prep_qkvT<<<cdiv((long)ATT * DD, TB), TB, 0, stream>>>(Wv_c, wvT[1]);
    prep_T<<<cdiv((long)ATT * ATT, TB), TB, 0, stream>>>(Wo_p, woT[0], ATT, ATT);
    prep_T<<<cdiv((long)ATT * ATT, TB), TB, 0, stream>>>(Wo_c, woT[1], ATT, ATT);
    prep_T<<<cdiv((long)CH * CH, TB), TB, 0, stream>>>(W_dag, wdagT, CH, CH);
    prep_T<<<cdiv((long)TS * ATT, TB), TB, 0, stream>>>(wp_w, wgT[0], TS, ATT);
    prep_T<<<cdiv((long)TS * ATT, TB), TB, 0, stream>>>(wc_w, wgT[1], TS, ATT);
    prep_cvt<<<cdiv((long)TS * TS, TB), TB, 0, stream>>>(emb_p, embb[0], (long)TS * TS);
    prep_cvt<<<cdiv((long)TS * TS, TB), TB, 0, stream>>>(emb_c, embb[1], (long)TS * TS);

    // ---- layernorm ----
    layernorm_kernel<<<cdiv((long)NN * 32, TB), TB, 0, stream>>>(node_states, ln_g, ln_b, x, xb);

    const int* srcs[2] = { pair_prem, pair_conj };
    const int* tgts[2] = { pair_conj, pair_prem };
    const float* gbias[2] = { wp_b, wc_b };

    const int blocks_m  = NN / 32;          // one block per 32-row tile
    const int shA_qkv   = 32 * DD * 2;      // LDS bytes for A tile, K = 256
    const int shA_wo    = 32 * ATT * 2;     // K = 512
    const int shA_wdag  = 32 * CH * 2;      // K = 768

    for (int d = 0; d < 2; ++d) {
        // Q/K/V projections (bf16 out), A tile staged via TDM
        wmma_gemm_bf16<5, false, DD, ATT><<<blocks_m, TB, shA_qkv, stream>>>(
            xb, wqT[d], nullptr, Qb, nullptr, nullptr, nullptr, NN);
        wmma_gemm_bf16<5, false, DD, ATT><<<blocks_m, TB, shA_qkv, stream>>>(
            xb, wkT[d], nullptr, Kb, nullptr, nullptr, nullptr, NN);
        wmma_gemm_bf16<5, false, DD, ATT><<<blocks_m, TB, shA_qkv, stream>>>(
            xb, wvT[d], nullptr, Vb, nullptr, nullptr, nullptr, NN);
        // reset segment-softmax state
        fill_f32<<<cdiv((long)NN * HH, TB), TB, 0, stream>>>(mbuf, -3.0e38f, (long)NN * HH);
        fill_f32<<<cdiv((long)NN * HH, TB), TB, 0, stream>>>(den, 0.0f, (long)NN * HH);
        fill_f32<<<cdiv((long)NN * ATT, TB), TB, 0, stream>>>(agg, 0.0f, (long)NN * ATT);
        // scores -> segment max -> exp/denominator -> scatter aggregation
        score_kernel<<<cdiv((long)EE * HH, TB), TB, 0, stream>>>(Qb, Kb, srcs[d], tgts[d], sc, mbuf);
        exp_kernel<<<cdiv((long)EE * HH, TB), TB, 0, stream>>>(srcs[d], sc, mbuf, den);
        agg_kernel<<<cdiv((long)EE * ATT, TB), TB, 0, stream>>>(srcs[d], tgts[d], sc, den, Vb, agg);
        cvt_f2b<<<cdiv((long)NN * ATT, TB), TB, 0, stream>>>(agg, aggb, (long)NN * ATT);
        // gate = sigmoid(emb[type] @ W + b)   (gathered-A WMMA GEMM, K=64, no LDS)
        wmma_gemm_bf16<1, true, TS, ATT><<<blocks_m, TB, 0, stream>>>(
            embb[d], wgT[d], gatebuf, nullptr, gbias[d], nullptr, node_types, NN);
        // comb (dir0: overwrite, dir1: accumulate) = gate * (agg @ Wo)
        if (d == 0)
            wmma_gemm_bf16<2, false, ATT, ATT><<<blocks_m, TB, shA_wo, stream>>>(
                aggb, woT[0], comb, nullptr, nullptr, gatebuf, nullptr, NN);
        else
            wmma_gemm_bf16<3, false, ATT, ATT><<<blocks_m, TB, shA_wo, stream>>>(
                aggb, woT[1], comb, nullptr, nullptr, gatebuf, nullptr, NN);
    }

    // ---- phase 2: pooled = tanh([x|comb] @ W_dag + b) ----
    build_h<<<cdiv((long)NN * CH, TB), TB, 0, stream>>>(x, comb, hb);
    wmma_gemm_bf16<4, false, CH, CH><<<blocks_m, TB, shA_wdag, stream>>>(
        hb, wdagT, pooled, nullptr, b_dag, nullptr, nullptr, NN);

    // ---- segment-max pooling + gather into output [P, 2*CH] ----
    fill_f32<<<cdiv((long)PP * 2 * CH, TB), TB, 0, stream>>>(out, -2.0f, (long)PP * 2 * CH);
    fill_f32<<<cdiv((long)PCC * CH, TB), TB, 0, stream>>>(csets, -2.0f, (long)PCC * CH);
    pool_max<<<cdiv((long)MPP * CH, TB), TB, 0, stream>>>(pooled, pool_p_nod, pool_p_seg, out, 2 * CH, MPP);
    pool_max<<<cdiv((long)MCC * CH, TB), TB, 0, stream>>>(pooled, pool_c_nod, pool_c_seg, csets, CH, MCC);
    gather_conj<<<cdiv((long)PP * CH, TB), TB, 0, stream>>>(csets, prem2conj, out);
}
